// HybridTransformerBlockQuantum_65481071395581
// MI455X (gfx1250) — matrix operand order repaired and checked by reading, not compiled, not run on hardware
//
#include <hip/hip_runtime.h>

// ---------------------------------------------------------------------------
// Hybrid transformer block for MI455X (gfx1250), wave32 + WMMA f16->f32.
// GEMMs: v_wmma_f32_16x16x32_f16, pre-transposed weights, double-buffered LDS
// with async global->LDS copies (ASYNCcnt) when the toolchain exposes them.
// Attention: flash-style online softmax (no materialized scores).
// ---------------------------------------------------------------------------

typedef _Float16 v16h __attribute__((ext_vector_type(16)));
typedef _Float16 v8h  __attribute__((ext_vector_type(8)));
typedef float    v8f  __attribute__((ext_vector_type(8)));
typedef int      vi4  __attribute__((vector_size(16)));

#define B_  4
#define S_  2048
#define E_  1024
#define H_  16
#define DK_ 64
#define FF_ 4096
#define NT_ (B_ * S_)

#if defined(__has_builtin)
#if __has_builtin(__builtin_amdgcn_global_load_async_to_lds_b128) && \
    __has_builtin(__builtin_amdgcn_s_wait_asynccnt)
#define USE_ASYNC_LDS 1
#endif
#endif

struct Frag { union { v16h f; v8h h[2]; _Float16 e[16]; }; };

static __device__ inline v8f wmma16(const Frag& a, const Frag& b, v8f c) {
  // D = A(16x32 f16) * B(32x16 f16) + C(16x16 f32)
  return __builtin_amdgcn_wmma_f32_16x16x32_f16(
      /*neg_a=*/false, a.f, /*neg_b=*/false, b.f,
      /*c_mod=*/(short)0, c, /*reuse_a=*/false, /*reuse_b=*/false);
}

#ifdef USE_ASYNC_LDS
// async copy of 16 bytes/lane global -> LDS (global_load_async_to_lds_b128).
// Builtin prototype (from toolchain diagnostic): (v4i addrspace(1)*,
// v4i addrspace(3)*, imm offset, imm cpol).
static __device__ inline void async_ld16(const _Float16* g, _Float16* l) {
  __builtin_amdgcn_global_load_async_to_lds_b128(
      (__attribute__((address_space(1))) vi4*)(uintptr_t)g,
      (__attribute__((address_space(3))) vi4*)(unsigned int)(uintptr_t)l,
      0, 0);
}
#endif

// --------------------------- elementwise kernels ---------------------------

__global__ void cvt_f16(const float* __restrict__ in, _Float16* __restrict__ out, int n) {
  int i = blockIdx.x * 256 + threadIdx.x;
  if (i < n) out[i] = (_Float16)in[i];
}

// f32 [K][N] -> f16 [N][K] transposed convert (tiled through LDS).
__global__ __launch_bounds__(256) void cvt_f16_T(const float* __restrict__ in,
                                                 _Float16* __restrict__ out,
                                                 int K, int N) {
  __shared__ float t[32][33];
  const int tx = threadIdx.x & 31, ty = threadIdx.x >> 5;   // 32 x 8
  const int k0 = blockIdx.y * 32, n0 = blockIdx.x * 32;
#pragma unroll
  for (int j = 0; j < 4; ++j)
    t[ty + j * 8][tx] = in[(size_t)(k0 + ty + j * 8) * N + n0 + tx];
  __syncthreads();
#pragma unroll
  for (int j = 0; j < 4; ++j)
    out[(size_t)(n0 + ty + j * 8) * K + k0 + tx] = (_Float16)t[tx][ty + j * 8];
}

// quantum head: cos(x + theta), theta broadcast per 64-wide wire group
__global__ void cos_embed(const float* __restrict__ x, const float* __restrict__ theta,
                          _Float16* __restrict__ out, int n) {
  int i = blockIdx.x * 256 + threadIdx.x;
  if (i < n) out[i] = (_Float16)__cosf(x[i] + theta[i & (DK_ - 1)]);
}

// ------------------------------- WMMA GEMM ---------------------------------
// C[M,N] = epilogue( A[M,K] @ Bt[N,K]^T )  -- A f16 row-major, Bt f16 N-major
// (pre-transposed weights). Block tile 128x128, 256 threads = 8 waves (4x2),
// wave tile 32x64, K-step 32, 2-stage LDS double buffer.

__global__ __launch_bounds__(256) void gemm_f16_wmma(
    const _Float16* __restrict__ A, const _Float16* __restrict__ Bt,
    const float* __restrict__ bias, float* __restrict__ outF,
    _Float16* __restrict__ outH,
    int M, int N, int K, float scale, int relu, int accum) {
  __shared__ _Float16 As[2][128 * 32];   // [row][k]
  __shared__ _Float16 Bs[2][128 * 32];   // [n][k]

  const int tid  = threadIdx.x;
  const int lane = tid & 31;
  const int wv   = tid >> 5;
  const int m16  = lane & 15;
  const int hlf  = lane >> 4;
  const int wm   = wv >> 1;           // 0..3
  const int wn   = wv & 1;            // 0..1
  const int m0   = blockIdx.y * 128;
  const int n0   = blockIdx.x * 128;

  const v8f zero = {};
  v8f acc[2][4];
#pragma unroll
  for (int mt = 0; mt < 2; ++mt)
#pragma unroll
    for (int nt = 0; nt < 4; ++nt) acc[mt][nt] = zero;

  // cooperative 128x32 tile copy: 2 threads/row, 16 halfs (32B) per thread
  const int crow = tid >> 1, ccol = (tid & 1) * 16;
  const _Float16* aSrc = A  + (size_t)(m0 + crow) * K + ccol;
  const _Float16* bSrc = Bt + (size_t)(n0 + crow) * K + ccol;
  const int ldst = crow * 32 + ccol;

  auto issue = [&](int st, int k0) {
#ifdef USE_ASYNC_LDS
    async_ld16(aSrc + k0,     &As[st][ldst]);
    async_ld16(aSrc + k0 + 8, &As[st][ldst + 8]);
    async_ld16(bSrc + k0,     &Bs[st][ldst]);
    async_ld16(bSrc + k0 + 8, &Bs[st][ldst + 8]);
#else
    *(v8h*)&As[st][ldst]     = *(const v8h*)(aSrc + k0);
    *(v8h*)&As[st][ldst + 8] = *(const v8h*)(aSrc + k0 + 8);
    *(v8h*)&Bs[st][ldst]     = *(const v8h*)(bSrc + k0);
    *(v8h*)&Bs[st][ldst + 8] = *(const v8h*)(bSrc + k0 + 8);
#endif
  };

  const int nIt = K >> 5;
  issue(0, 0);

  for (int i = 0; i < nIt; ++i) {
    __syncthreads();              // retire prev-iter reads of the other stage
    if (i + 1 < nIt) {
      issue((i + 1) & 1, (i + 1) << 5);
#ifdef USE_ASYNC_LDS
      __builtin_amdgcn_s_wait_asynccnt(4);   // stage i copies (older 4) done
#endif
    } else {
#ifdef USE_ASYNC_LDS
      __builtin_amdgcn_s_wait_asynccnt(0);
#endif
    }
    __syncthreads();              // stage i visible to all waves

    const _Float16* as = As[i & 1];
    const _Float16* bs = Bs[i & 1];

    Frag a[2], b[4];
#pragma unroll
    for (int mt = 0; mt < 2; ++mt) {   // A frag 16x32: lane m=m16
      int r = wm * 32 + mt * 16 + m16;
      a[mt].h[0] = *(const v8h*)&as[r * 32 + hlf * 8];
      a[mt].h[1] = *(const v8h*)&as[r * 32 + 16 + hlf * 8];
    }
#pragma unroll
    for (int nt = 0; nt < 4; ++nt) {   // B frag 32x16: lane n=m16, k=hlf*8..+7, 16+hlf*8..+7
      int c = wn * 64 + nt * 16 + m16;
      b[nt].h[0] = *(const v8h*)&bs[c * 32 + hlf * 8];
      b[nt].h[1] = *(const v8h*)&bs[c * 32 + 16 + hlf * 8];
    }
#pragma unroll
    for (int mt = 0; mt < 2; ++mt)
#pragma unroll
      for (int nt = 0; nt < 4; ++nt)
        acc[mt][nt] = wmma16(a[mt], b[nt], acc[mt][nt]);
  }

  // Epilogue. C layout: VGPR r, lane -> (m = r + 8*hlf, n = m16)
#pragma unroll
  for (int mt = 0; mt < 2; ++mt)
#pragma unroll
    for (int nt = 0; nt < 4; ++nt)
#pragma unroll
      for (int r = 0; r < 8; ++r) {
        int row = m0 + wm * 32 + mt * 16 + r + 8 * hlf;
        int col = n0 + wn * 64 + nt * 16 + m16;
        float v = acc[mt][nt][r] * scale;
        if (bias) v += bias[col];
        if (relu) v = fmaxf(v, 0.0f);
        size_t idx = (size_t)row * N + col;
        if (accum && outF) v += outF[idx];
        if (outF) outF[idx] = v;
        if (outH) outH[idx] = (_Float16)v;
      }
}

// --------------------------- flash attention -------------------------------
// One block = one (b,h) and 128 query rows. 8 waves; each wave owns 16 rows.
// Keys processed in chunks of 64 with online softmax. q is pre-scaled 1/8.

__global__ __launch_bounds__(256) void flash_attn(
    const _Float16* __restrict__ q, const _Float16* __restrict__ k,
    const _Float16* __restrict__ v, _Float16* __restrict__ ctx) {
  __shared__ _Float16 Kl[64 * 64];        // [key][dk]  row-major (QK^T B-frags)
  __shared__ _Float16 Vt[64 * 64];        // [dk][key]  transposed (PV B-frags)
  __shared__ _Float16 Ps[8][16 * 64];     // per-wave P staging, C->A relayout

  const int tid  = threadIdx.x;
  const int lane = tid & 31;
  const int wv   = tid >> 5;
  const int m16  = lane & 15;
  const int hlf  = lane >> 4;
  const int bh   = blockIdx.y;
  const int bb   = bh >> 4;
  const int hh   = bh & 15;
  const int q0   = blockIdx.x * 128;

  // Q fragments (16 rows x 64 dk = 2 A-frags), loaded straight from global
  Frag qa[2];
  {
    const size_t qrow = (size_t)bb * S_ + q0 + wv * 16 + m16;
#pragma unroll
    for (int kk = 0; kk < 2; ++kk) {
      const _Float16* src = q + qrow * E_ + hh * 64 + kk * 32;
      qa[kk].h[0] = *(const v8h*)(src + hlf * 8);
      qa[kk].h[1] = *(const v8h*)(src + 16 + hlf * 8);
    }
  }

  const v8f zero = {};
  v8f o[4];
#pragma unroll
  for (int dt = 0; dt < 4; ++dt) o[dt] = zero;
  float mrow[8], lrow[8];
#pragma unroll
  for (int r = 0; r < 8; ++r) { mrow[r] = -1e30f; lrow[r] = 0.0f; }

  const int lr = tid >> 2, lc = (tid & 3) * 16;   // cooperative 64x64 loaders

  for (int j = 0; j < S_; j += 64) {
    __syncthreads();
    {
      const _Float16* ks = k + ((size_t)bb * S_ + j + lr) * E_ + hh * 64 + lc;
#ifdef USE_ASYNC_LDS
      async_ld16(ks,     &Kl[lr * 64 + lc]);
      async_ld16(ks + 8, &Kl[lr * 64 + lc + 8]);
#else
      *(v8h*)&Kl[lr * 64 + lc]     = *(const v8h*)ks;
      *(v8h*)&Kl[lr * 64 + lc + 8] = *(const v8h*)(ks + 8);
#endif
      Frag t;   // V chunk transposed into LDS (overlaps the K async copy)
      const _Float16* vs = v + ((size_t)bb * S_ + j + lr) * E_ + hh * 64 + lc;
      t.h[0] = *(const v8h*)vs;
      t.h[1] = *(const v8h*)(vs + 8);
#pragma unroll
      for (int i = 0; i < 16; ++i) Vt[(lc + i) * 64 + lr] = t.e[i];
#ifdef USE_ASYNC_LDS
      __builtin_amdgcn_s_wait_asynccnt(0);
#endif
    }
    __syncthreads();

    // S = Q K^T  (4 key-tiles of 16, K-dim = DK = 64 -> 2 WMMAs each)
    v8f s[4];
#pragma unroll
    for (int nt = 0; nt < 4; ++nt) {
      s[nt] = zero;
#pragma unroll
      for (int kk = 0; kk < 2; ++kk) {
        Frag kb;
        kb.h[0] = *(const v8h*)&Kl[(nt * 16 + m16) * 64 + kk * 32 + hlf * 8];
        kb.h[1] = *(const v8h*)&Kl[(nt * 16 + m16) * 64 + kk * 32 + 16 + hlf * 8];
        s[nt] = wmma16(qa[kk], kb, s[nt]);
      }
    }

    // Online softmax per row (row = r + 8*hlf, spread over 16 lanes)
#pragma unroll
    for (int r = 0; r < 8; ++r) {
      float x0 = s[0][r], x1 = s[1][r], x2 = s[2][r], x3 = s[3][r];
      float mx = fmaxf(fmaxf(x0, x1), fmaxf(x2, x3));
#pragma unroll
      for (int off = 1; off < 16; off <<= 1) mx = fmaxf(mx, __shfl_xor(mx, off, 16));
      float mnew = fmaxf(mrow[r], mx);
      float sc   = __expf(mrow[r] - mnew);
      float p0 = __expf(x0 - mnew), p1 = __expf(x1 - mnew);
      float p2 = __expf(x2 - mnew), p3 = __expf(x3 - mnew);
      float rs = p0 + p1 + p2 + p3;
#pragma unroll
      for (int off = 1; off < 16; off <<= 1) rs += __shfl_xor(rs, off, 16);
      lrow[r] = lrow[r] * sc + rs;
      mrow[r] = mnew;
#pragma unroll
      for (int dt = 0; dt < 4; ++dt) o[dt][r] *= sc;
      _Float16* pp = &Ps[wv][(r + 8 * hlf) * 64];
      pp[ 0 + m16] = (_Float16)p0;
      pp[16 + m16] = (_Float16)p1;
      pp[32 + m16] = (_Float16)p2;
      pp[48 + m16] = (_Float16)p3;
    }

    // O += P @ V  (wave-private staging; same-wave DS ops stay in order)
#pragma unroll
    for (int kk = 0; kk < 2; ++kk) {
      Frag pa;
      pa.h[0] = *(const v8h*)&Ps[wv][m16 * 64 + kk * 32 + hlf * 8];
      pa.h[1] = *(const v8h*)&Ps[wv][m16 * 64 + kk * 32 + 16 + hlf * 8];
#pragma unroll
      for (int dt = 0; dt < 4; ++dt) {
        Frag vb;
        vb.h[0] = *(const v8h*)&Vt[(dt * 16 + m16) * 64 + kk * 32 + hlf * 8];
        vb.h[1] = *(const v8h*)&Vt[(dt * 16 + m16) * 64 + kk * 32 + 16 + hlf * 8];
        o[dt] = wmma16(pa, vb, o[dt]);
      }
    }
  }

  // normalize and store ctx as f16 [token][E] (heads interleaved back)
#pragma unroll
  for (int dt = 0; dt < 4; ++dt)
#pragma unroll
    for (int r = 0; r < 8; ++r) {
      size_t row = (size_t)bb * S_ + q0 + wv * 16 + r + 8 * hlf;
      int col = hh * 64 + dt * 16 + m16;
      ctx[row * E_ + col] = (_Float16)(o[dt][r] / lrow[r]);
    }
}

// ------------------------------ layernorm ----------------------------------
// One block per token row (E = 1024, 4 elems/thread). Optional fused residual
// input (base + exs*extra); writes f32 and/or f16.

__global__ __launch_bounds__(256) void ln_fused(
    const float* __restrict__ base, const float* __restrict__ extra, float exs,
    const float* __restrict__ g, const float* __restrict__ bta,
    float* __restrict__ outF, _Float16* __restrict__ outH) {
  __shared__ float red[256];
  const int tid = threadIdx.x;
  const size_t row = blockIdx.x;
  float vals[4];
  float s1 = 0.0f, s2 = 0.0f;
#pragma unroll
  for (int kq = 0; kq < 4; ++kq) {
    int c = kq * 256 + tid;
    size_t i = row * E_ + c;
    float vv = base[i];
    if (extra) vv += exs * extra[i];
    vals[kq] = vv; s1 += vv; s2 += vv * vv;
  }
  red[tid] = s1; __syncthreads();
  for (int off = 128; off > 0; off >>= 1) { if (tid < off) red[tid] += red[tid + off]; __syncthreads(); }
  float mu = red[0] * (1.0f / E_); __syncthreads();
  red[tid] = s2; __syncthreads();
  for (int off = 128; off > 0; off >>= 1) { if (tid < off) red[tid] += red[tid + off]; __syncthreads(); }
  float var = red[0] * (1.0f / E_) - mu * mu;
  float rstd = rsqrtf(var + 1e-5f);
#pragma unroll
  for (int kq = 0; kq < 4; ++kq) {
    int c = kq * 256 + tid;
    size_t i = row * E_ + c;
    float y = (vals[kq] - mu) * rstd * g[c] + bta[c];
    if (outF) outF[i] = y;
    if (outH) outH[i] = (_Float16)y;
  }
}

// ------------------------------- launcher ----------------------------------

extern "C" void kernel_launch(void* const* d_in, const int* in_sizes, int n_in,
                              void* d_out, int out_size, void* d_ws, size_t ws_size,
                              hipStream_t stream) {
  (void)in_sizes; (void)n_in; (void)out_size; (void)ws_size;
  const float* x     = (const float*)d_in[0];
  const float* Wq    = (const float*)d_in[1];
  const float* Wk    = (const float*)d_in[2];
  const float* Wv    = (const float*)d_in[3];
  const float* Wo    = (const float*)d_in[4];
  const float* theta = (const float*)d_in[5];
  const float* Wcq   = (const float*)d_in[6];
  const float* ln1g  = (const float*)d_in[7];
  const float* ln1b  = (const float*)d_in[8];
  const float* W1    = (const float*)d_in[9];
  const float* b1    = (const float*)d_in[10];
  const float* W2    = (const float*)d_in[11];
  const float* b2    = (const float*)d_in[12];
  const float* ln2g  = (const float*)d_in[13];
  const float* ln2b  = (const float*)d_in[14];
  float* out = (float*)d_out;

  // workspace carve-out (256B aligned)
  char* p = (char*)d_ws;
  auto take = [&](size_t elems, size_t esz) -> void* {
    void* q = (void*)p;
    p += (elems * esz + 255) & ~(size_t)255;
    return q;
  };
  _Float16* x_h   = (_Float16*)take((size_t)NT_ * E_, 2);
  _Float16* WqT   = (_Float16*)take((size_t)E_ * E_, 2);   // [N][K] transposed
  _Float16* WkT   = (_Float16*)take((size_t)E_ * E_, 2);
  _Float16* WvT   = (_Float16*)take((size_t)E_ * E_, 2);
  _Float16* WoT   = (_Float16*)take((size_t)E_ * E_, 2);
  _Float16* WcqT  = (_Float16*)take((size_t)E_ * E_, 2);
  _Float16* W1T   = (_Float16*)take((size_t)E_ * FF_, 2);  // [FF][E]
  _Float16* W2T   = (_Float16*)take((size_t)FF_ * E_, 2);  // [E][FF]
  _Float16* q_h   = (_Float16*)take((size_t)NT_ * E_, 2);
  _Float16* k_h   = (_Float16*)take((size_t)NT_ * E_, 2);
  _Float16* v_h   = (_Float16*)take((size_t)NT_ * E_, 2);
  _Float16* ctx_h = (_Float16*)take((size_t)NT_ * E_, 2);
  _Float16* cos_h = (_Float16*)take((size_t)NT_ * E_, 2);
  _Float16* x1_h  = (_Float16*)take((size_t)NT_ * E_, 2);
  _Float16* h1_h  = (_Float16*)take((size_t)NT_ * FF_, 2);
  float*    cls   = (float*)take((size_t)NT_ * E_, 4);   // classical + quantum
  float*    r2    = (float*)take((size_t)NT_ * E_, 4);   // x1 + ffn residual

  const int nXE = NT_ * E_;          // 8388608
  dim3 blk(256);

  // conversions: x plain, weights transposed to [N][K] f16
  cvt_f16<<<(nXE + 255) / 256, blk, 0, stream>>>(x, x_h, nXE);
  {
    dim3 gT(E_ / 32, E_ / 32);                 // (32, 32)
    cvt_f16_T<<<gT, blk, 0, stream>>>(Wq,  WqT,  E_, E_);
    cvt_f16_T<<<gT, blk, 0, stream>>>(Wk,  WkT,  E_, E_);
    cvt_f16_T<<<gT, blk, 0, stream>>>(Wv,  WvT,  E_, E_);
    cvt_f16_T<<<gT, blk, 0, stream>>>(Wo,  WoT,  E_, E_);
    cvt_f16_T<<<gT, blk, 0, stream>>>(Wcq, WcqT, E_, E_);
    cvt_f16_T<<<dim3(FF_ / 32, E_ / 32), blk, 0, stream>>>(W1, W1T, E_, FF_);
    cvt_f16_T<<<dim3(E_ / 32, FF_ / 32), blk, 0, stream>>>(W2, W2T, FF_, E_);
  }

  dim3 gE(E_ / 128, NT_ / 128);      // (8, 64)
  dim3 gF(FF_ / 128, NT_ / 128);     // (32, 64)

  // Q/K/V projections (Q pre-scaled by 1/sqrt(DK) = 0.125)
  gemm_f16_wmma<<<gE, blk, 0, stream>>>(x_h, WqT, nullptr, nullptr, q_h, NT_, E_, E_, 0.125f, 0, 0);
  gemm_f16_wmma<<<gE, blk, 0, stream>>>(x_h, WkT, nullptr, nullptr, k_h, NT_, E_, E_, 1.0f,   0, 0);
  gemm_f16_wmma<<<gE, blk, 0, stream>>>(x_h, WvT, nullptr, nullptr, v_h, NT_, E_, E_, 1.0f,   0, 0);

  // quantum feature map
  cos_embed<<<(nXE + 255) / 256, blk, 0, stream>>>(x, theta, cos_h, nXE);

  // flash attention -> ctx
  flash_attn<<<dim3(S_ / 128, B_ * H_), blk, 0, stream>>>(q_h, k_h, v_h, ctx_h);

  // classical = ctx @ Wo ; then cls += cos @ Wcq (fused accumulate)
  gemm_f16_wmma<<<gE, blk, 0, stream>>>(ctx_h, WoT,  nullptr, cls, nullptr, NT_, E_, E_, 1.0f, 0, 0);
  gemm_f16_wmma<<<gE, blk, 0, stream>>>(cos_h, WcqT, nullptr, cls, nullptr, NT_, E_, E_, 1.0f, 0, 1);

  // x1 = LN(x + 0.5*(classical+quantum)); r2 = x1 (f32), x1_h (f16)
  ln_fused<<<NT_, blk, 0, stream>>>(x, cls, 0.5f, ln1g, ln1b, r2, x1_h);

  // FFN: h1 = relu(x1@W1 + b1); r2 += h1@W2 + b2
  gemm_f16_wmma<<<gF, blk, 0, stream>>>(x1_h, W1T, b1, nullptr, h1_h, NT_, FF_, E_, 1.0f, 1, 0);
  gemm_f16_wmma<<<gE, blk, 0, stream>>>(h1_h, W2T, b2, r2, nullptr, NT_, E_, FF_, 1.0f, 0, 1);

  // out = LN(x1 + ffn)
  ln_fused<<<NT_, blk, 0, stream>>>(r2, nullptr, 0.0f, ln2g, ln2b, out, nullptr);
}
